// GraphSync2_46669114638603
// MI455X (gfx1250) — compile-verified, matrix operand order checked by reading, not run on hardware
//
#include <hip/hip_runtime.h>

typedef __attribute__((ext_vector_type(16))) _Float16 v16h;
typedef __attribute__((ext_vector_type(8)))  float    v8f;
typedef __attribute__((ext_vector_type(4)))  float    v4f;

#define LRELU_ALPHA 0.2f

__device__ __forceinline__ float dleaky(float x)   { return x > 0.0f ? x : LRELU_ALPHA * x; }
__device__ __forceinline__ float dsigmoid(float x) { return 1.0f / (1.0f + expf(-x)); }

// ---------------------------------------------------------------------------
// FAST PATH: row-major A (unit k-stride), row-major B (unit n-stride),
// M % 16 == 0, N % 16 == 0, Kd % 32 == 0. No guards at all.
// One wave computes one 16x16 tile. A fragment = 4x global_load_b128.
// ---------------------------------------------------------------------------
__global__ void wmma_gemm_fast(const float* __restrict__ A, long lda, long saB,
                               const float* __restrict__ B, long ldb, long sbB,
                               const float* __restrict__ bias, float* __restrict__ C,
                               long ldc, long scB, int Kd)
{
    const int lane  = threadIdx.x;             // 0..31
    const int tn = blockIdx.x, tm = blockIdx.y, bz = blockIdx.z;
    const int l16   = lane & 15;
    const int khalf = (lane >> 4) * 8;         // 0 for lanes 0-15, 8 for 16-31

    const float* Ab = A + (long)bz * saB + (long)(tm * 16 + l16) * lda;  // row base
    const float* Bb = B + (long)bz * sbB + (tn * 16 + l16);              // column base

    v8f acc = {};
    for (int k0 = 0; k0 < Kd; k0 += 32) {
        const float* pa = Ab + k0 + khalf;
        const float* pb = Bb + (long)(k0 + khalf) * ldb;
        __builtin_prefetch(pa + 32, 0, 1);                 // next A tile
        __builtin_prefetch(pb + 32 * ldb, 0, 1);           // next B tile
        // A fragment: halves [0..7] -> K base+0..7, [8..15] -> K base+16..23
        const v4f a0 = *(const v4f*)(pa);
        const v4f a1 = *(const v4f*)(pa + 4);
        const v4f a2 = *(const v4f*)(pa + 16);
        const v4f a3 = *(const v4f*)(pa + 20);
        v16h av, bv;
        av[0]=(_Float16)a0[0]; av[1]=(_Float16)a0[1]; av[2]=(_Float16)a0[2]; av[3]=(_Float16)a0[3];
        av[4]=(_Float16)a1[0]; av[5]=(_Float16)a1[1]; av[6]=(_Float16)a1[2]; av[7]=(_Float16)a1[3];
        av[8]=(_Float16)a2[0]; av[9]=(_Float16)a2[1]; av[10]=(_Float16)a2[2]; av[11]=(_Float16)a2[3];
        av[12]=(_Float16)a3[0]; av[13]=(_Float16)a3[1]; av[14]=(_Float16)a3[2]; av[15]=(_Float16)a3[3];
#pragma unroll
        for (int e = 0; e < 16; ++e) {
            const long krel = ((e < 8) ? 0 : 16) + (e & 7);   // constant after unroll
            bv[e] = (_Float16)pb[krel * ldb];
        }
        acc = __builtin_amdgcn_wmma_f32_16x16x32_f16(
                  false, av, false, bv, (short)0, acc, false, false);
    }

    const int ncol = tn * 16 + l16;
    const float badd = (bias != nullptr) ? bias[ncol] : 0.0f;   // uniform branch
    float* Cb = C + (long)bz * scB + (long)(tm * 16 + khalf) * ldc + ncol;
#pragma unroll
    for (int r = 0; r < 8; ++r)
        Cb[(long)r * ldc] = acc[r] + badd;
}

// ---------------------------------------------------------------------------
// GENERIC PATH: arbitrary strides + ragged M/N/K handled by clamped addresses
// and 0/1 mask multiplies (v_cndmask, never EXEC branches / predicated loads).
// ---------------------------------------------------------------------------
__global__ void wmma_gemm_gen(const float* __restrict__ A, const float* __restrict__ B,
                              const float* __restrict__ bias, float* __restrict__ C,
                              int M, int N, int Kd,
                              long saM, long saK, long saB,
                              long sbK, long sbN, long sbB,
                              long scM, long scB)
{
    const int lane = threadIdx.x;
    const int tn = blockIdx.x, tm = blockIdx.y, bz = blockIdx.z;
    const int l16   = lane & 15;
    const int khalf = (lane >> 4) * 8;

    const int arow = tm * 16 + l16;
    const int bcol = tn * 16 + l16;
    const int ar = (arow < M) ? arow : (M - 1);       // clamped (always in range)
    const int bc = (bcol < N) ? bcol : (N - 1);
    const float amask = (arow < M) ? 1.0f : 0.0f;
    const float bmask = (bcol < N) ? 1.0f : 0.0f;

    const float* Ab = A + (long)bz * saB + (long)ar * saM;
    const float* Bb = B + (long)bz * sbB + (long)bc * sbN;

    v8f acc = {};
    for (int k0 = 0; k0 < Kd; k0 += 32) {
        v16h av, bv;
#pragma unroll
        for (int e = 0; e < 16; ++e) {
            const int kk = k0 + ((e < 8) ? 0 : 16) + khalf + (e & 7);
            const int kc = (kk < Kd) ? kk : (Kd - 1);             // clamp
            const float km = (kk < Kd) ? 1.0f : 0.0f;
            const float fa = Ab[(long)kc * saK] * (km * amask);   // unconditional load
            const float fb = Bb[(long)kc * sbK] * (km * bmask);
            av[e] = (_Float16)fa;
            bv[e] = (_Float16)fb;
        }
        acc = __builtin_amdgcn_wmma_f32_16x16x32_f16(
                  false, av, false, bv, (short)0, acc, false, false);
    }

    const float badd = (bias != nullptr && bcol < N) ? bias[bcol] : 0.0f;
    const int mbase = tm * 16 + khalf;
#pragma unroll
    for (int r = 0; r < 8; ++r) {
        const int m = mbase + r;
        if (m < M && bcol < N)
            C[(long)bz * scB + (long)m * scM + bcol] = acc[r] + badd;
    }
}

// ---------------------------------------------------------------------------
__global__ void vec_add(const float* __restrict__ a, const float* __restrict__ b,
                        float* __restrict__ o, int n)
{
    int i = blockIdx.x * blockDim.x + threadIdx.x;
    if (i < n) o[i] = a[i] + b[i];
}

// graph = softmax(score / sqrt(c), axis=1)  (softmax over i, per (b,j) column)
__global__ void graph_softmax(const float* __restrict__ score,
                              float* __restrict__ graph, float scale)
{
    const int b = blockIdx.x >> 6;
    const int j = blockIdx.x & 63;
    const int i = threadIdx.x;                    // 0..63
    __shared__ float red[64];
    float v = score[(long)b * 4096 + i * 64 + j] * scale;
    red[i] = v; __syncthreads();
    for (int off = 32; off > 0; off >>= 1) {
        if (i < off) red[i] = fmaxf(red[i], red[i + off]);
        __syncthreads();
    }
    float mx = red[0]; __syncthreads();
    float e = expf(v - mx);
    red[i] = e; __syncthreads();
    for (int off = 32; off > 0; off >>= 1) {
        if (i < off) red[i] += red[i + off];
        __syncthreads();
    }
    graph[(long)b * 4096 + i * 64 + j] = e / red[0];
}

__device__ float softmax64(float v, float* red, int j)
{
    __syncthreads();
    red[j] = v; __syncthreads();
    for (int off = 32; off > 0; off >>= 1) {
        if (j < off) red[j] = fmaxf(red[j], red[j + off]);
        __syncthreads();
    }
    float mx = red[0]; __syncthreads();
    float e = expf(v - mx);
    red[j] = e; __syncthreads();
    for (int off = 32; off > 0; off >>= 1) {
        if (j < off) red[j] += red[j + off];
        __syncthreads();
    }
    float s = red[0]; __syncthreads();
    return e / s;
}

// Per (sensor n, row i): 3-scale pairwise scores -> softmax_j -> mul -> t_h row
__global__ void temporal_kernel(const float* __restrict__ x,     // (256,64,32)
                                const float* __restrict__ vi1, const float* __restrict__ vj1, // (16384,64)
                                const float* __restrict__ vi2, const float* __restrict__ vj2, // (16384,32)
                                const float* __restrict__ vi3, const float* __restrict__ vj3, // (16384,16)
                                const float* __restrict__ a1, const float* __restrict__ a2,
                                const float* __restrict__ a3,
                                const float* __restrict__ mul_w, const float* __restrict__ mul_b,
                                float* __restrict__ t_h)          // (256,64,32)
{
    const int i = blockIdx.x;      // 0..63 window row
    const int n = blockIdx.y;      // 0..255 sensor
    const int j = threadIdx.x;     // 0..63 window col
    __shared__ float vi1s[64], vi2s[32], vi3s[16], red[64], muls[64];
    const long ri = (long)n * 64 + i;
    const long rj = (long)n * 64 + j;
    vi1s[j] = vi1[ri * 64 + j];               // bias lin_b pre-folded into vi1
    if (j < 32) vi2s[j] = vi2[ri * 32 + j];   // bias lin2_b pre-folded into vi2
    if (j < 16) vi3s[j] = vi3[ri * 16 + j];   // bias lin3_b pre-folded into vi3
    __syncthreads();

    float e1 = 0.f, e2 = 0.f, e3 = 0.f;
    for (int e = 0; e < 64; ++e) e1 += dleaky(vi1s[e] + vj1[rj * 64 + e]) * a1[e];
    for (int e = 0; e < 32; ++e) e2 += dleaky(vi2s[e] + vj2[rj * 32 + e]) * a2[e];
    for (int e = 0; e < 16; ++e) e3 += dleaky(vi3s[e] + vj3[rj * 16 + e]) * a3[e];

    const float s1 = softmax64(e1, red, j);
    const float s2 = softmax64(e2, red, j);
    const float s3 = softmax64(e3, red, j);

    float m = dleaky(s1 * mul_w[0] + s2 * mul_w[1] + s3 * mul_w[2] + mul_b[0]);
    // upper-triangular log-decay weights: log1p(64-(j-i))/log1p(64) for i<=j
    const float wm = (i <= j) ? (logf(1.0f + (float)(64 - (j - i))) / logf(65.0f)) : 0.0f;
    muls[j] = m * wm;
    __syncthreads();

    if (j < 32) {  // t_h[n,i,d] = sigmoid(sum_t mul[t] * v1[n,t,d])
        float acc = 0.f;
        for (int t = 0; t < 64; ++t)
            acc += muls[t] * x[((long)n * 64 + t) * 32 + j];
        t_h[ri * 32 + j] = dsigmoid(acc);
    }
}

// One block per sequence; Whh staged in LDS (64 KB of the 320 KB WGP pool)
__global__ void lstm_kernel(const float* __restrict__ xg,   // (256,64,256) = x@Wih+bih+bhh
                            const float* __restrict__ Whh,  // (64,256)
                            float* __restrict__ hout)       // (256,64,64)
{
    const int n = blockIdx.x;
    const int j = threadIdx.x;          // 0..63 hidden index
    __shared__ float Whh_s[64 * 256];
    __shared__ float h_s[64];
    for (int idx = j; idx < 64 * 256; idx += 64) Whh_s[idx] = Whh[idx];
    h_s[j] = 0.0f;
    float c = 0.0f;
    __syncthreads();
    for (int t = 0; t < 64; ++t) {
        const float* xr = xg + ((long)n * 64 + t) * 256;
        float gi = xr[j], gf = xr[64 + j], gg = xr[128 + j], go = xr[192 + j];
        for (int k = 0; k < 64; ++k) {
            const float hk = h_s[k];
            const float* wr = Whh_s + k * 256;
            gi += hk * wr[j];       gf += hk * wr[64 + j];
            gg += hk * wr[128 + j]; go += hk * wr[192 + j];
        }
        const float iv = dsigmoid(gi), fv = dsigmoid(gf);
        const float gv = tanhf(gg),    ov = dsigmoid(go);
        c = fv * c + iv * gv;
        const float h = ov * tanhf(c);
        __syncthreads();
        h_s[j] = h;
        hout[((long)n * 64 + t) * 64 + j] = h;
        __syncthreads();
    }
}

// hn[:, :, 1:] += hr[:, :, :-1];  relu
__global__ void gnn_combine(const float* __restrict__ hn2, const float* __restrict__ hr,
                            float* __restrict__ hnrelu)
{
    const long row = blockIdx.x;           // (b,k,l) row, 0..16383
    const int  d   = threadIdx.x;          // 0..63
    const int  l   = (int)(row & 63);
    const long idx = row * 64 + d;
    float v = hn2[idx];
    if (l > 0) v += hr[idx - 64];
    hnrelu[idx] = fmaxf(v, 0.0f);
}

// cost = 1 - gram; global min/max reduce (single block)
__global__ void cost_minmax(const float* __restrict__ raw, float* __restrict__ cost,
                            float* __restrict__ mm)
{
    const int tid = threadIdx.x;           // 256 threads
    float lmin = 1e30f, lmax = -1e30f;
    for (int idx = tid; idx < 16384; idx += 256) {
        const float v = 1.0f - raw[idx];
        cost[idx] = v;
        lmin = fminf(lmin, v); lmax = fmaxf(lmax, v);
    }
    __shared__ float rmin[256], rmax[256];
    rmin[tid] = lmin; rmax[tid] = lmax; __syncthreads();
    for (int off = 128; off > 0; off >>= 1) {
        if (tid < off) {
            rmin[tid] = fminf(rmin[tid], rmin[tid + off]);
            rmax[tid] = fmaxf(rmax[tid], rmax[tid + off]);
        }
        __syncthreads();
    }
    if (tid == 0) { mm[0] = rmin[0]; mm[1] = rmax[0]; }
}

__global__ void cost_thresh(float* __restrict__ cost, const float* __restrict__ mm)
{
    const int idx = blockIdx.x * 256 + threadIdx.x;
    if (idx < 16384) {
        const float thr = mm[0] + 0.2f * (mm[1] - mm[0]);
        cost[idx] = fmaxf(cost[idx] - thr, 0.0f);
    }
}

// IPOT (beta=0.5, 50 iters) on 64x64, one block per batch, fully LDS-resident
__global__ void ipot_kernel(const float* __restrict__ cost, float* __restrict__ ot)
{
    const int b = blockIdx.x;
    const int i = threadIdx.x;             // 0..63, one row per thread
    __shared__ float As[4096], Ts[4096], Qs[4096];
    __shared__ float ss[64], ds[64], red[64];
    const float* Cb = cost + (long)b * 4096;
    for (int j = 0; j < 64; ++j) {
        const float cv = Cb[i * 64 + j];
        As[i * 64 + j] = fmaxf(expf(-2.0f * cv), 1e-8f);   // exp(-C/beta)
        Ts[i * 64 + j] = 1.0f / 64.0f;
    }
    ss[i] = 1.0f / 64.0f;
    __syncthreads();
    for (int it = 0; it < 50; ++it) {
        float qs = 0.0f;
        for (int j = 0; j < 64; ++j) {
            const float q = As[i * 64 + j] * Ts[i * 64 + j];
            Qs[i * 64 + j] = q;
            qs += q * ss[j];
        }
        ds[i] = 1.0f / (qs * 64.0f + 1e-8f);
        __syncthreads();
        float t = 0.0f;
        for (int k = 0; k < 64; ++k) t += Qs[k * 64 + i] * ds[k];
        const float snew = 1.0f / (t * 64.0f + 1e-8f);
        __syncthreads();
        ss[i] = snew;
        __syncthreads();
        for (int j = 0; j < 64; ++j)
            Ts[i * 64 + j] = ds[i] * Qs[i * 64 + j] * ss[j];
        __syncthreads();
    }
    float p = 0.0f;
    for (int j = 0; j < 64; ++j) p += Cb[i * 64 + j] * Ts[i * 64 + j];
    red[i] = p; __syncthreads();
    for (int off = 32; off > 0; off >>= 1) {
        if (i < off) red[i] += red[i + off];
        __syncthreads();
    }
    if (i == 0) ot[b] = red[0];
}

// ---------------------------------------------------------------------------
extern "C" void kernel_launch(void* const* d_in, const int* in_sizes, int n_in,
                              void* d_out, int out_size, void* d_ws, size_t ws_size,
                              hipStream_t stream)
{
    (void)in_sizes; (void)n_in; (void)out_size; (void)ws_size;
    const float* x      = (const float*)d_in[0];   // (4,64,64,32)
    const float* wq     = (const float*)d_in[1];
    const float* bq     = (const float*)d_in[2];
    const float* wk     = (const float*)d_in[3];
    const float* bk     = (const float*)d_in[4];
    const float* lin_w  = (const float*)d_in[5];   // (64,64)
    const float* lin_b  = (const float*)d_in[6];
    const float* a1     = (const float*)d_in[7];
    const float* el2_w  = (const float*)d_in[8];   // (32,16)
    const float* el2_b  = (const float*)d_in[9];
    const float* lin2_w = (const float*)d_in[10];  // (32,32)
    const float* lin2_b = (const float*)d_in[11];
    const float* a2     = (const float*)d_in[12];
    const float* el3_w  = (const float*)d_in[13];  // (32,8)
    const float* el3_b  = (const float*)d_in[14];
    const float* lin3_w = (const float*)d_in[15];  // (16,16)
    const float* lin3_b = (const float*)d_in[16];
    const float* a3     = (const float*)d_in[17];
    const float* mul_w  = (const float*)d_in[18];
    const float* mul_b  = (const float*)d_in[19];
    const float* Wih    = (const float*)d_in[20];  // (32,256)
    const float* Whh    = (const float*)d_in[21];  // (64,256)
    const float* bih    = (const float*)d_in[22];
    const float* bhh    = (const float*)d_in[23];
    const float* gn_w   = (const float*)d_in[24];  // (64,64)
    const float* gn_b   = (const float*)d_in[25];
    const float* gr_w   = (const float*)d_in[26];
    const float* g2_w   = (const float*)d_in[27];
    const float* g2_b   = (const float*)d_in[28];

    // ---- workspace carve (floats) ----
    float* ws = (float*)d_ws;
    float* q       = ws; ws += 524288;   // (256,2048)
    float* kbuf    = ws; ws += 524288;   // (256,2048)
    float* score   = ws; ws += 16384;    // (4,64,64)
    float* graph   = ws; ws += 16384;    // (4,64,64)
    float* vi1     = ws; ws += 1048576;  // (16384,64)
    float* vj1     = ws; ws += 1048576;
    float* v2      = ws; ws += 262144;   // (16384,16)
    float* vi2     = ws; ws += 524288;   // (16384,32)
    float* vj2     = ws; ws += 524288;
    float* v3      = ws; ws += 131072;   // (16384,8)
    float* vi3     = ws; ws += 262144;   // (16384,16)
    float* vj3     = ws; ws += 262144;
    float* xg      = ws; ws += 4194304;  // (16384,256)
    float* bsum    = ws; ws += 256;
    float* hlstm   = ws; ws += 1048576;  // (256,64,64) == (4,64,64,64)
    float* hagg    = ws; ws += 1048576;  // (4,64,4096)
    float* hn2     = ws; ws += 1048576;
    float* hr      = ws; ws += 1048576;
    float* hnrelu  = ws; ws += 1048576;
    float* costraw = ws; ws += 16384;
    float* cost    = ws; ws += 16384;
    float* mm      = ws; ws += 2;

    float* out_h  = (float*)d_out;              // (4,64,64,64) = 1048576
    float* out_th = out_h + 1048576;            // (4,64,64,32) = 524288
    float* out_ot = out_h + 1572864;            // (4,)

    // fast path: row-major A/B, M,N %16==0, Kd %32==0
    auto gemmF = [&](const float* Ap, long lda, long saB,
                     const float* Bp, long ldb, long sbB,
                     const float* biasp, float* Cp, long ldc, long scB,
                     int M, int N, int Kd, int batch) {
        dim3 g((unsigned)(N / 16), (unsigned)(M / 16), (unsigned)batch);
        wmma_gemm_fast<<<g, dim3(32, 1, 1), 0, stream>>>(Ap, lda, saB, Bp, ldb, sbB,
                                                         biasp, Cp, ldc, scB, Kd);
    };
    // generic path: arbitrary strides, ragged sizes
    auto gemmG = [&](const float* Ap, const float* Bp, const float* biasp, float* Cp,
                     int M, int N, int Kd, int batch,
                     long saM, long saK, long saB,
                     long sbK, long sbN, long sbB,
                     long scM, long scB) {
        dim3 g((unsigned)((N + 15) / 16), (unsigned)((M + 15) / 16), (unsigned)batch);
        wmma_gemm_gen<<<g, dim3(32, 1, 1), 0, stream>>>(Ap, Bp, biasp, Cp, M, N, Kd,
                                                        saM, saK, saB, sbK, sbN, sbB, scM, scB);
    };

    // --- sensor-graph attention: q,k projections (2048x2048) + score + softmax
    gemmF(x, 2048, 0,  wq, 2048, 0,  bq, q,    2048, 0,  256, 2048, 2048, 1);
    gemmF(x, 2048, 0,  wk, 2048, 0,  bk, kbuf, 2048, 0,  256, 2048, 2048, 1);
    // score[b] = q[b] @ k[b].reshape(c,K) : B element (a,j) = kbuf[b, a*64+j]
    gemmF(q, 2048, 131072,  kbuf, 64, 131072,  nullptr, score, 64, 4096,  64, 64, 2048, 4);
    graph_softmax<<<dim3(256), dim3(64), 0, stream>>>(score, graph, 1.0f / sqrtf(2048.0f));

    // --- temporal attention projections (bias of pairwise "pre" folded into vi)
    gemmF(x, 32, 0,  lin_w,           64, 0,  lin_b,   vi1, 64, 0,  16384, 64, 32, 1);
    gemmF(x, 32, 0,  lin_w + 32 * 64, 64, 0,  nullptr, vj1, 64, 0,  16384, 64, 32, 1);
    gemmF(x, 32, 0,  el2_w,           16, 0,  el2_b,   v2,  16, 0,  16384, 16, 32, 1);
    gemmG(v2, lin2_w,           lin2_b,  vi2, 16384, 32, 16, 1, 16, 1, 0, 32, 1, 0, 32, 0);
    gemmG(v2, lin2_w + 16 * 32, nullptr, vj2, 16384, 32, 16, 1, 16, 1, 0, 32, 1, 0, 32, 0);
    gemmG(x,  el3_w,            el3_b,   v3,  16384,  8, 32, 1, 32, 1, 0,  8, 1, 0,  8, 0);
    gemmG(v3, lin3_w,           lin3_b,  vi3, 16384, 16,  8, 1,  8, 1, 0, 16, 1, 0, 16, 0);
    gemmG(v3, lin3_w + 8 * 16,  nullptr, vj3, 16384, 16,  8, 1,  8, 1, 0, 16, 1, 0, 16, 0);
    temporal_kernel<<<dim3(64, 256), dim3(64), 0, stream>>>(
        x, vi1, vj1, vi2, vj2, vi3, vj3, a1, a2, a3, mul_w, mul_b, out_th);

    // --- LSTM: xg = v1 @ Wih + (bih+bhh), then LDS-resident recurrence
    vec_add<<<dim3(1), dim3(256), 0, stream>>>(bih, bhh, bsum, 256);
    gemmF(x, 32, 0,  Wih, 256, 0,  bsum, xg, 256, 0,  16384, 256, 32, 1);
    lstm_kernel<<<dim3(256), dim3(64), 0, stream>>>(xg, Whh, hlstm);

    // --- GNN: hagg[b,j,:] = sum_k graph[b,k,j] * h[b,k,:]   (A = graph^T -> generic)
    gemmG(graph, hlstm, nullptr, hagg, 64, 4096, 64, 4,
          1, 64, 4096,   4096, 1, 262144,   4096, 262144);
    gemmF(hagg,  64, 0,  gn_w, 64, 0,  gn_b,    hn2, 64, 0,  16384, 64, 64, 1);
    gemmF(hlstm, 64, 0,  gr_w, 64, 0,  nullptr, hr,  64, 0,  16384, 64, 64, 1);
    gnn_combine<<<dim3(16384), dim3(64), 0, stream>>>(hn2, hr, hnrelu);
    gemmF(hnrelu, 64, 0,  g2_w, 64, 0,  g2_b, out_h, 64, 0,  16384, 64, 64, 1);

    // --- cost = 1 - hf @ hf^T (B transposed -> generic), threshold, IPOT
    gemmG(out_h, out_h, nullptr, costraw, 64, 64, 4096, 4,
          4096, 1, 262144,   1, 4096, 262144,   64, 4096);
    cost_minmax<<<dim3(1), dim3(256), 0, stream>>>(costraw, cost, mm);
    cost_thresh<<<dim3(64), dim3(256), 0, stream>>>(cost, mm);
    ipot_kernel<<<dim3(4), dim3(64), 0, stream>>>(cost, out_ot);
}
